// DeepCrossing_87179246174820
// MI455X (gfx1250) — matrix operand order repaired and checked by reading, not compiled, not run on hardware
//
#include <hip/hip_runtime.h>
#include <hip/hip_bf16.h>
#include <math.h>
#include <stdint.h>

typedef _Float16 v16h __attribute__((ext_vector_type(16)));
typedef _Float16 v8h  __attribute__((ext_vector_type(8)));
typedef _Float16 v4h  __attribute__((ext_vector_type(4)));
typedef _Float16 v2h  __attribute__((ext_vector_type(2)));
typedef float    v8f  __attribute__((ext_vector_type(8)));
typedef int      v4i  __attribute__((ext_vector_type(4)));

#define BM 128
#define BN 128
#define BK 32
#define A_STRIDE 40    // halfs: 32 + 8 pad  (row offset = 80B, 16B aligned)
#define B_STRIDE 136   // halfs: 128 + 8 pad (row offset = 272B, 16B aligned)

union V16U { v16h v; v8h h[2]; };

// Async global->LDS staging (ASYNCcnt path) if the toolchain exposes it.
#if __has_builtin(__builtin_amdgcn_global_load_async_to_lds_b128) && \
    __has_builtin(__builtin_amdgcn_s_wait_asynccnt)
#define USE_ASYNC_LDS 1
#else
#define USE_ASYNC_LDS 0
#endif

#if USE_ASYNC_LDS
typedef __attribute__((address_space(1))) v4i gas_v4i;   // global int4
typedef __attribute__((address_space(3))) v4i las_v4i;   // LDS int4
__device__ __forceinline__ void async_copy_b128(const void* g, void* l) {
  // generic->global is bit-identical; generic LDS addr truncates to the
  // 32-bit LDS offset (ISA 10.2 aperture rules), so integer casts are exact.
  __builtin_amdgcn_global_load_async_to_lds_b128(
      (gas_v4i*)(uintptr_t)g, (las_v4i*)(uint32_t)(uintptr_t)l, 0, 0);
}
#endif

// -------------------------------------------------------------------------
// 1) Embedding gather + sum-pool: x[B,F,L] int64 -> feat fp32 + featH f16.
// -------------------------------------------------------------------------
__global__ void embed_pool_kernel(const long long* __restrict__ x,
                                  const float* __restrict__ emb,
                                  float* __restrict__ feat,
                                  _Float16* __restrict__ featH) {
  const int lane = threadIdx.x & 31;
  const int wave = threadIdx.x >> 5;
  const int p = blockIdx.x * 8 + wave;        // (b,f) pair index, B*F = 106496
  const int b = p / 26;
  const int f = p % 26;
  const long long* ids = x + ((long long)b * 26 + f) * 8;
  float2 acc = make_float2(0.f, 0.f);
#pragma unroll
  for (int l = 0; l < 8; ++l) {
    const long long id = ids[l];
    const float2 v = *(const float2*)(emb + (size_t)id * 64 + lane * 2);
    acc.x += v.x;
    acc.y += v.y;
  }
  const size_t o = (size_t)b * 1664 + f * 64 + lane * 2;
  *(float2*)(feat + o) = acc;
  v2h h = { (_Float16)acc.x, (_Float16)acc.y };
  *(v2h*)(featH + o) = h;
}

// -------------------------------------------------------------------------
// 2) fp32 -> f16 weight conversion (weights then stay resident in 192MB L2)
// -------------------------------------------------------------------------
__global__ void f32_to_f16_kernel(const float4* __restrict__ in,
                                  v4h* __restrict__ out, int n4) {
  const int i = blockIdx.x * blockDim.x + threadIdx.x;
  if (i < n4) {
    const float4 v = in[i];
    v4h h = { (_Float16)v.x, (_Float16)v.y, (_Float16)v.z, (_Float16)v.w };
    out[i] = h;
  }
}

// -------------------------------------------------------------------------
// 3) Double-buffered WMMA GEMM, all-f16 operands, fp32 accumulate:
//      out = relu( A[M,K]f16 @ Bw[K,N]f16 + bias (+ res fp32) )
//    Staging of tile k+1 is issued (async DMA to LDS when available) before
//    the WMMAs of tile k; one s_wait_asynccnt + barrier per iteration.
// -------------------------------------------------------------------------
template <bool HAS_RES, bool WRITE_F32, bool WRITE_F16>
__global__ __launch_bounds__(256)
void gemm_wmma_kernel(const _Float16* __restrict__ A,
                      const _Float16* __restrict__ Bw,
                      const float* __restrict__ bias,
                      const float* __restrict__ res,
                      float* __restrict__ C,
                      _Float16* __restrict__ Ch,
                      int M, int N, int K) {
  __shared__ __align__(16) _Float16 Alds[2][BM * A_STRIDE];
  __shared__ __align__(16) _Float16 Blds[2][BK * B_STRIDE];

  const int tid  = threadIdx.x;
  const int lane = tid & 31;
  const int wave = tid >> 5;
  const int wm = wave & 3;   // 4 waves down M: 32 rows each
  const int wn = wave >> 2;  // 2 waves across N: 64 cols each
  const int m0 = blockIdx.x * BM;
  const int n0 = blockIdx.y * BN;

  // staging geometry: each thread moves 2 x 16B for A and 2 x 16B for B
  const int aRow = tid >> 2;          // 0..63 (+64 for second slot)
  const int aCol = (tid & 3) << 3;    // 0,8,16,24
  const int bRow = tid >> 4;          // 0..15 (+16 for second slot)
  const int bCol = (tid & 15) << 3;   // 0..120

  const _Float16* aG0 = A + (size_t)(m0 + aRow) * K + aCol;
  const _Float16* aG1 = A + (size_t)(m0 + aRow + 64) * K + aCol;
  const _Float16* bG0 = Bw + (size_t)bRow * N + n0 + bCol;
  const _Float16* bG1 = Bw + (size_t)(bRow + 16) * N + n0 + bCol;

  v8f acc[2][4];
#pragma unroll
  for (int mi = 0; mi < 2; ++mi)
#pragma unroll
    for (int ni = 0; ni < 4; ++ni)
      acc[mi][ni] = {};

  const int kTiles = K / BK;

  auto stage = [&](int buf, int kt2) {
    const _Float16* ga0 = aG0 + (size_t)kt2 * BK;
    const _Float16* ga1 = aG1 + (size_t)kt2 * BK;
    const _Float16* gb0 = bG0 + (size_t)kt2 * BK * N;
    const _Float16* gb1 = bG1 + (size_t)kt2 * BK * N;
    _Float16* la0 = &Alds[buf][aRow * A_STRIDE + aCol];
    _Float16* la1 = &Alds[buf][(aRow + 64) * A_STRIDE + aCol];
    _Float16* lb0 = &Blds[buf][bRow * B_STRIDE + bCol];
    _Float16* lb1 = &Blds[buf][(bRow + 16) * B_STRIDE + bCol];
#if USE_ASYNC_LDS
    async_copy_b128(ga0, la0);
    async_copy_b128(ga1, la1);
    async_copy_b128(gb0, lb0);
    async_copy_b128(gb1, lb1);
#else
    const v8h a0 = *(const v8h*)ga0;
    const v8h a1 = *(const v8h*)ga1;
    const v8h b0 = *(const v8h*)gb0;
    const v8h b1 = *(const v8h*)gb1;
    *(v8h*)la0 = a0;
    *(v8h*)la1 = a1;
    *(v8h*)lb0 = b0;
    *(v8h*)lb1 = b1;
#endif
  };

  // prologue: stage tile 0 into buffer 0
  stage(0, 0);
#if USE_ASYNC_LDS
  __builtin_amdgcn_s_wait_asynccnt(0);
#endif
  __syncthreads();

  for (int kt = 0; kt < kTiles; ++kt) {
    const int cur = kt & 1;
    const bool haveNext = (kt + 1 < kTiles);

    // issue staging of tile kt+1 into the alternate buffer (overlaps WMMAs)
    if (haveNext) stage(cur ^ 1, kt + 1);
    if (kt + 2 < kTiles) {  // L2 prefetch for the streaming A operand
      __builtin_prefetch(aG0 + (size_t)(kt + 2) * BK, 0, 3);
      __builtin_prefetch(aG1 + (size_t)(kt + 2) * BK, 0, 3);
    }

    // fragment loads per ISA 7.12.2 wave32 layouts
    V16U af[2], bf[4];
    const int kbase = (lane >> 4) << 3;  // lanes 0-15: K 0-7/16-23; 16-31: K 8-15/24-31
#pragma unroll
    for (int mi = 0; mi < 2; ++mi) {
      const _Float16* p = &Alds[cur][(wm * 32 + mi * 16 + (lane & 15)) * A_STRIDE + kbase];
      af[mi].h[0] = *(const v8h*)(p);
      af[mi].h[1] = *(const v8h*)(p + 16);
    }
#pragma unroll
    for (int ni = 0; ni < 4; ++ni) {
      const _Float16* p = &Blds[cur][kbase * B_STRIDE + wn * 64 + ni * 16 + (lane & 15)];  // B: lane = N col, elements = K
#pragma unroll
      for (int s = 0; s < 8; ++s) {
        bf[ni].v[s] = p[s * B_STRIDE];
        bf[ni].v[8 + s] = p[(16 + s) * B_STRIDE];
      }
    }

#pragma unroll
    for (int mi = 0; mi < 2; ++mi)
#pragma unroll
      for (int ni = 0; ni < 4; ++ni)
        acc[mi][ni] = __builtin_amdgcn_wmma_f32_16x16x32_f16(
            false, af[mi].v, false, bf[ni].v, (short)0, acc[mi][ni], false, false);

    if (haveNext) {
#if USE_ASYNC_LDS
      __builtin_amdgcn_s_wait_asynccnt(0);
#endif
      __syncthreads();
    }
  }

  // epilogue: bias (+ fp32 residual) + ReLU; fp32 and/or f16 stores
#pragma unroll
  for (int mi = 0; mi < 2; ++mi) {
#pragma unroll
    for (int ni = 0; ni < 4; ++ni) {
      const int col = n0 + wn * 64 + ni * 16 + (lane & 15);
      const float bv = bias[col];
      const int rbase = m0 + wm * 32 + mi * 16 + ((lane >> 4) << 3);
#pragma unroll
      for (int r = 0; r < 8; ++r) {
        const int row = rbase + r;
        float v = acc[mi][ni][r] + bv;
        if (HAS_RES) v += res[(size_t)row * N + col];
        v = v > 0.f ? v : 0.f;
        if (WRITE_F32) C[(size_t)row * N + col] = v;
        if (WRITE_F16) Ch[(size_t)row * N + col] = (_Float16)v;
      }
    }
  }
}

// -------------------------------------------------------------------------
// 4) Final linear [1664 -> 1] + sigmoid; one wave per row, shfl reduction.
// -------------------------------------------------------------------------
__global__ void final_sigmoid_kernel(const float* __restrict__ feat,
                                     const float* __restrict__ w,
                                     const float* __restrict__ b,
                                     float* __restrict__ out) {
  const int lane = threadIdx.x & 31;
  const int wave = threadIdx.x >> 5;
  const int row = blockIdx.x * 8 + wave;
  const float* fr = feat + (size_t)row * 1664;
  float s = 0.f;
#pragma unroll 4
  for (int i = lane; i < 1664; i += 32) s += fr[i] * w[i];
#pragma unroll
  for (int off = 16; off > 0; off >>= 1) s += __shfl_down(s, off, 32);
  if (lane == 0) out[row] = 1.0f / (1.0f + __expf(-(s + b[0])));
}

// -------------------------------------------------------------------------
extern "C" void kernel_launch(void* const* d_in, const int* in_sizes, int n_in,
                              void* d_out, int out_size, void* d_ws, size_t ws_size,
                              hipStream_t stream) {
  const long long* x  = (const long long*)d_in[0];   // reference: int64 indices
  const float* emb    = (const float*)d_in[1];
  const float* w1[3]  = {(const float*)d_in[2],  (const float*)d_in[6],  (const float*)d_in[10]};
  const float* b1[3]  = {(const float*)d_in[3],  (const float*)d_in[7],  (const float*)d_in[11]};
  const float* w2[3]  = {(const float*)d_in[4],  (const float*)d_in[8],  (const float*)d_in[12]};
  const float* b2[3]  = {(const float*)d_in[5],  (const float*)d_in[9],  (const float*)d_in[13]};
  const float* lin_w  = (const float*)d_in[14];
  const float* lin_b  = (const float*)d_in[15];
  float* out = (float*)d_out;

  constexpr int Bsz = 4096, F = 26, IN = 1664;
  constexpr int H[3] = {1024, 1024, 512};

  // workspace carve-up (256B aligned):
  //   feat fp32 27.3MB, featH f16 13.6MB, hH f16 8.4MB, f16 weights ~17MB
  char* ws = (char*)d_ws;
  size_t off = 0;
  auto alloc = [&](size_t bytes) -> void* {
    void* p = ws + off;
    off = (off + bytes + 255) & ~(size_t)255;
    return p;
  };
  float*    feat  = (float*)alloc((size_t)Bsz * IN * sizeof(float));
  _Float16* featH = (_Float16*)alloc((size_t)Bsz * IN * sizeof(_Float16));
  _Float16* hH    = (_Float16*)alloc((size_t)Bsz * 1024 * sizeof(_Float16));
  _Float16* w1h[3];
  _Float16* w2h[3];
  for (int i = 0; i < 3; ++i) {
    w1h[i] = (_Float16*)alloc((size_t)IN * H[i] * sizeof(_Float16));
    w2h[i] = (_Float16*)alloc((size_t)H[i] * IN * sizeof(_Float16));
  }

  // 1) embedding gather + pool (fp32 + f16 copies)
  embed_pool_kernel<<<(Bsz * F) / 8, 256, 0, stream>>>(x, emb, feat, featH);

  // 2) weight conversion fp32 -> f16
  for (int i = 0; i < 3; ++i) {
    const int n = IN * H[i];  // same count both directions
    f32_to_f16_kernel<<<(n / 4 + 255) / 256, 256, 0, stream>>>(
        (const float4*)w1[i], (v4h*)w1h[i], n / 4);
    f32_to_f16_kernel<<<(n / 4 + 255) / 256, 256, 0, stream>>>(
        (const float4*)w2[i], (v4h*)w2h[i], n / 4);
  }

  // 3) three residual blocks:
  //    hH = relu(featH @ w1 + b1)              (f16-only output, feeds GEMM2)
  //    feat = relu(hH @ w2 + b2 + feat),  featH = f16(feat)
  for (int i = 0; i < 3; ++i) {
    dim3 g1(Bsz / BM, H[i] / BN);
    gemm_wmma_kernel<false, false, true><<<g1, 256, 0, stream>>>(
        featH, w1h[i], b1[i], nullptr, nullptr, hH, Bsz, H[i], IN);
    dim3 g2(Bsz / BM, IN / BN);
    gemm_wmma_kernel<true, true, true><<<g2, 256, 0, stream>>>(
        hH, w2h[i], b2[i], feat, feat, featH, Bsz, IN, H[i]);
  }

  // 4) final linear + sigmoid -> out[B,1]
  final_sigmoid_kernel<<<Bsz / 8, 256, 0, stream>>>(feat, lin_w, lin_b, out);
}
